// InferenceMachine_52046413693093
// MI455X (gfx1250) — compile-verified, matrix operand order checked by reading, not run on hardware
//
#include <hip/hip_runtime.h>
#include <hip/hip_bf16.h>
#include <math.h>

typedef __bf16 bf16;
typedef __attribute__((ext_vector_type(16))) __bf16 v16bf;
typedef __attribute__((ext_vector_type(8)))  float  v8f;

#define B_ROWS 8192
#define HDIM   256
#define NPOL   4096
#define KPAD   160   // 135 padded up to multiple of 32

#define LSTAT_COLS  64
#define LSTAT_ROWS  64
#define LSTAT_ITERS (NPOL / LSTAT_COLS)

__device__ __forceinline__ bf16 f2bf(float f) {
    union { float f; unsigned u; } x; x.f = f;
    unsigned r = x.u + 0x7FFFu + ((x.u >> 16) & 1u);      // round-to-nearest-even
    union { unsigned short s; bf16 b; } y; y.s = (unsigned short)(r >> 16);
    return y.b;
}
__device__ __forceinline__ float bf2f(bf16 b) {
    union { unsigned short s; bf16 b; } y; y.b = b;
    union { unsigned u; float f; } x; x.u = ((unsigned)y.s) << 16;
    return x.f;
}

union Frag { uint4 u[2]; v16bf v; };

// one 16-byte async global->LDS transfer; INST_OFFSET applies to BOTH addresses
#define ASYNC_B128(OFF, LDSA, GPTR)                                          \
    asm volatile("global_load_async_to_lds_b128 %0, %1, off offset:" #OFF    \
                 :: "v"(LDSA), "v"(GPTR) : "memory")

#define WAIT_ASYNC0() asm volatile("s_wait_asynccnt 0x0" ::: "memory")

// ---------------- prep: build padded input matrix k (B x 160, bf16) -------------
__global__ void prep_k(const float* __restrict__ pf, const float* __restrict__ mb,
                       const float* __restrict__ ks, const float* __restrict__ mo,
                       const float* __restrict__ ps, bf16* __restrict__ kpad) {
    int idx = blockIdx.x * 256 + threadIdx.x;
    if (idx >= B_ROWS * KPAD) return;
    int b = idx / KPAD, c = idx % KPAD;
    float v = 0.f;
    if (c < 131)       v = pf[b * 131 + c];
    else if (c == 131) v = mb[b];
    else if (c == 132) v = ks[b];
    else if (c == 133) v = mo[b];
    else if (c == 134) v = ps[b];
    kpad[idx] = f2bf(v);
}

// ---- prep: transpose fp32 weight (Ksrc x N) -> bf16 Wt (N x Kdst), zero-pad K ----
__global__ void prep_wt(const float* __restrict__ W, bf16* __restrict__ Wt,
                        int Ksrc, int Kdst, int N) {
    int idx = blockIdx.x * 256 + threadIdx.x;
    if (idx >= N * Kdst) return;
    int n = idx / Kdst, k = idx % Kdst;
    Wt[idx] = f2bf(k < Ksrc ? W[k * N + n] : 0.f);
}

// ---------------- dense layer: Out(bf16 B x 256) = act(A @ Wt^T + bias) ----------
// Wt is N(=256) x K row-major bf16. Block: 256 thr = 8 waves; 32 rows x 256 cols.
// K-loop software-pipelined: loads for step kk+32 issue before the WMMAs of kk.
template <int K, bool RELU>
__global__ __launch_bounds__(256, 2)
void gemm_bias_act(const bf16* __restrict__ A, const bf16* __restrict__ Wt,
                   const float* __restrict__ bias, bf16* __restrict__ Out) {
    const int w    = threadIdx.x >> 5;
    const int lane = threadIdx.x & 31;
    const int ln   = lane & 15, half = lane >> 4;
    const int r0   = blockIdx.x * 32 + (w >> 2) * 16;
    const int c0   = (w & 3) * 64;
    v8f acc[4] = {};
    const bf16* ap0 = A + (size_t)(r0 + ln) * K + half * 8;
    const bf16* wp0 = Wt + (size_t)(c0 + ln) * K + half * 8;

    Frag a[2], b[2][4];
    a[0].u[0] = *(const uint4*)ap0;
    a[0].u[1] = *(const uint4*)(ap0 + 16);
#pragma unroll
    for (int t = 0; t < 4; ++t) {
        const bf16* wp = wp0 + (size_t)(t * 16) * K;
        b[0][t].u[0] = *(const uint4*)wp;
        b[0][t].u[1] = *(const uint4*)(wp + 16);
    }
#pragma unroll
    for (int kk = 0; kk < K; kk += 32) {
        const int cur = (kk >> 5) & 1, nxt = cur ^ 1;
        if (kk + 32 < K) {
            const bf16* ap = ap0 + kk + 32;
            a[nxt].u[0] = *(const uint4*)ap;
            a[nxt].u[1] = *(const uint4*)(ap + 16);
#pragma unroll
            for (int t = 0; t < 4; ++t) {
                const bf16* wp = wp0 + (size_t)(t * 16) * K + kk + 32;
                b[nxt][t].u[0] = *(const uint4*)wp;
                b[nxt][t].u[1] = *(const uint4*)(wp + 16);
            }
        }
#pragma unroll
        for (int t = 0; t < 4; ++t)
            acc[t] = __builtin_amdgcn_wmma_f32_16x16x32_bf16(
                false, a[cur].v, false, b[cur][t].v, (short)0, acc[t], false, false);
    }
#pragma unroll
    for (int t = 0; t < 4; ++t) {
        int n = c0 + t * 16 + ln;
        float bv = bias[n];
#pragma unroll
        for (int i = 0; i < 8; ++i) {
            float v = acc[t][i] + bv;
            if (RELU) v = fmaxf(v, 0.f);
            Out[(size_t)(r0 + i + half * 8) * HDIM + n] = f2bf(v);
        }
    }
}

// -------- policy stats: stream h3 @ Wp^T + bp, online row max & sum-exp → lse ----
// Block: 512 thr = 16 waves, 64 rows. Wave w: row group w>>2 (16 rows), col tile
// w&3 within the 64-col panel. Weight panel (64 cols x 256 K, 32KB bf16) is
// double-buffered in LDS via async global->LDS DMA (ASYNCcnt). Softmax stats are
// kept PER LANE during the loop; cross-lane/wave merges happen once at the end.
__global__ __launch_bounds__(512, 1)
void logits_stats(const bf16* __restrict__ h3, const bf16* __restrict__ Wpt,
                  const float* __restrict__ bp, float* __restrict__ lse) {
    __shared__ __align__(16) bf16 ldsW[2][LSTAT_COLS * HDIM];   // 2 x 32KB
    __shared__ float smax[16][16];
    __shared__ float ssum[16][16];
    const int tid  = threadIdx.x;
    const int w    = tid >> 5;
    const int lane = tid & 31;
    const int ln   = lane & 15, half = lane >> 4;
    const int rg   = w >> 2;            // row group: 0..3
    const int ct   = w & 3;             // column tile 0..3 within 64-col panel
    const int r0   = blockIdx.x * LSTAT_ROWS + rg * 16;

    // async-copy assignment: each of 512 threads moves 64B of one Wp^T row
    const int crow = tid >> 3;          // 0..63: column (= row of Wp^T) in panel
    const int cseg = tid & 7;           // 0..7 : 64B segment of the 512B row

    // A fragments for this wave's 16-row block (K = 256) stay in registers.
    Frag afr[8];
    {
        const bf16* ap = h3 + (size_t)(r0 + ln) * HDIM + half * 8;
#pragma unroll
        for (int k8 = 0; k8 < 8; ++k8) {
            afr[k8].u[0] = *(const uint4*)(ap + k8 * 32);
            afr[k8].u[1] = *(const uint4*)(ap + k8 * 32 + 16);
        }
    }
    float rmax[8], rsum[8];
#pragma unroll
    for (int i = 0; i < 8; ++i) { rmax[i] = -3.0e38f; rsum[i] = 0.f; }

    // prefetch panel 0 into buffer 0
    {
        const char* gsrc = (const char*)Wpt + ((size_t)crow * HDIM + cseg * 32) * sizeof(bf16);
        unsigned ldsa = (unsigned)(uintptr_t)&ldsW[0][crow * HDIM + cseg * 32];
        ASYNC_B128(0, ldsa, gsrc); ASYNC_B128(16, ldsa, gsrc);
        ASYNC_B128(32, ldsa, gsrc); ASYNC_B128(48, ldsa, gsrc);
    }
    WAIT_ASYNC0();
    __syncthreads();

    for (int it = 0; it < LSTAT_ITERS; ++it) {
        const int cur = it & 1;
        if (it + 1 < LSTAT_ITERS) {   // kick off DMA of next panel into other buffer
            const char* gsrc = (const char*)Wpt +
                (((size_t)(it + 1) * LSTAT_COLS + crow) * HDIM + cseg * 32) * sizeof(bf16);
            unsigned ldsa = (unsigned)(uintptr_t)&ldsW[cur ^ 1][crow * HDIM + cseg * 32];
            ASYNC_B128(0, ldsa, gsrc); ASYNC_B128(16, ldsa, gsrc);
            ASYNC_B128(32, ldsa, gsrc); ASYNC_B128(48, ldsa, gsrc);
        }
        // 8 WMMAs over K=256, B-fragments double-buffered from LDS
        const bf16* lb = &ldsW[cur][(ct * 16 + ln) * HDIM + half * 8];
        Frag b[2];
        b[0].u[0] = *(const uint4*)lb;
        b[0].u[1] = *(const uint4*)(lb + 16);
        v8f acc = {};
#pragma unroll
        for (int k8 = 0; k8 < 8; ++k8) {
            if (k8 < 7) {
                b[(k8 + 1) & 1].u[0] = *(const uint4*)(lb + (k8 + 1) * 32);
                b[(k8 + 1) & 1].u[1] = *(const uint4*)(lb + (k8 + 1) * 32 + 16);
            }
            acc = __builtin_amdgcn_wmma_f32_16x16x32_bf16(
                false, afr[k8].v, false, b[k8 & 1].v, (short)0, acc, false, false);
        }
        // per-lane online softmax (this lane's column slice only; no shuffles here)
        const float bv = bp[it * LSTAT_COLS + ct * 16 + ln];
#pragma unroll
        for (int i = 0; i < 8; ++i) {
            float x = acc[i] + bv;
            float nm = fmaxf(rmax[i], x);
            rsum[i] = rsum[i] * __expf(rmax[i] - nm) + __expf(x - nm);
            rmax[i] = nm;
        }
        if (it + 1 < LSTAT_ITERS) WAIT_ASYNC0();
        __syncthreads();
    }

    // merge stats across the 16 lanes of each half-wave (log-sum-exp merge), once
#pragma unroll
    for (int i = 0; i < 8; ++i) {
        float m = rmax[i], s = rsum[i];
#pragma unroll
        for (int d = 1; d < 16; d <<= 1) {
            float om = __shfl_xor(m, d, 16);
            float os = __shfl_xor(s, d, 16);
            float nm = fmaxf(m, om);
            s = s * __expf(m - nm) + os * __expf(om - nm);
            m = nm;
        }
        rmax[i] = m; rsum[i] = s;
    }
    if (ln == 0) {
#pragma unroll
        for (int i = 0; i < 8; ++i) {
            smax[w][i + half * 8] = rmax[i];
            ssum[w][i + half * 8] = rsum[i];
        }
    }
    __syncthreads();
    // merge the 4 col-tile waves of each row group, write lse
    if (tid < 64) {
        const int g = tid >> 4, rl = tid & 15;
        float gm = -3.0e38f;
        for (int wv = 0; wv < 4; ++wv) gm = fmaxf(gm, smax[g * 4 + wv][rl]);
        float gs = 0.f;
        for (int wv = 0; wv < 4; ++wv) gs += ssum[g * 4 + wv][rl] * __expf(smax[g * 4 + wv][rl] - gm);
        lse[blockIdx.x * LSTAT_ROWS + g * 16 + rl] = gm + __logf(gs);
    }
}

// ---------------- eval head: tanh(h@Wh + Wenc[from] + Wenc[to+64] + be) ----------
__global__ void eval_head(const bf16* __restrict__ h3, const float* __restrict__ We,
                          const float* __restrict__ be, const int* __restrict__ from_sq,
                          const int* __restrict__ to_sq, float* __restrict__ evals) {
    const int w    = threadIdx.x >> 5;
    const int lane = threadIdx.x & 31;
    const int row  = blockIdx.x * 8 + w;
    const bf16* hr = h3 + (size_t)row * HDIM;
    float p = 0.f;
#pragma unroll
    for (int j = 0; j < 8; ++j) {
        int c = lane * 8 + j;
        p += bf2f(hr[c]) * We[c];
    }
    p += __shfl_xor(p, 1, 32);
    p += __shfl_xor(p, 2, 32);
    p += __shfl_xor(p, 4, 32);
    p += __shfl_xor(p, 8, 32);
    p += __shfl_xor(p, 16, 32);
    const float b0 = be[0];
#pragma unroll
    for (int q = 0; q < 2; ++q) {
        int mm = lane + q * 32;
        int fs = from_sq[row * 64 + mm];
        int ts = to_sq[row * 64 + mm];
        evals[(size_t)row * 64 + mm] = tanhf(p + We[HDIM + fs] + We[HDIM + 64 + ts] + b0);
    }
}

// -------- gather 64 logits per row, log-softmax via lse, clip, softmax(64) ------
__global__ __launch_bounds__(64)
void gather_probs(const bf16* __restrict__ h3, const bf16* __restrict__ Wpt,
                  const float* __restrict__ bp, const float* __restrict__ lse,
                  const int* __restrict__ from_sq, const int* __restrict__ to_sq,
                  float* __restrict__ probs) {
    __shared__ __align__(16) bf16 hs[HDIM];
    __shared__ float red[64];
    const int t = threadIdx.x;
    const int b = blockIdx.x;
    *(uint2*)(&hs[t * 4]) = *(const uint2*)(h3 + (size_t)b * HDIM + t * 4);
    __syncthreads();
    int fs = from_sq[b * 64 + t], ts = to_sq[b * 64 + t];
    int idx = fs * 64 + ts;
    const bf16* wr = Wpt + (size_t)idx * HDIM;   // contiguous 512B row of Wp^T
    float acc = 0.f;
#pragma unroll 8
    for (int j = 0; j < HDIM; ++j) acc += bf2f(hs[j]) * bf2f(wr[j]);
    float logp = acc + bp[idx] - lse[b];
    float c = fminf(10.f, fmaxf(-10.f, logp));
    red[t] = c; __syncthreads();
    for (int s = 32; s > 0; s >>= 1) { if (t < s) red[t] = fmaxf(red[t], red[t + s]); __syncthreads(); }
    float bmax = red[0]; __syncthreads();
    red[t] = __expf(c - bmax); __syncthreads();
    for (int s = 32; s > 0; s >>= 1) { if (t < s) red[t] += red[t + s]; __syncthreads(); }
    float bsum = red[0];
    probs[(size_t)b * 64 + t] = __expf(c - bmax) / bsum;
}

extern "C" void kernel_launch(void* const* d_in, const int* in_sizes, int n_in,
                              void* d_out, int out_size, void* d_ws, size_t ws_size,
                              hipStream_t stream) {
    (void)in_sizes; (void)n_in; (void)out_size; (void)ws_size;
    const float* pf  = (const float*)d_in[0];
    const float* mb  = (const float*)d_in[1];
    const float* ksf = (const float*)d_in[2];
    const float* mo  = (const float*)d_in[3];
    const float* ps  = (const float*)d_in[4];
    const int*   from_sq = (const int*)d_in[5];
    const int*   to_sq   = (const int*)d_in[6];
    const float* W1 = (const float*)d_in[7];
    const float* b1 = (const float*)d_in[8];
    const float* W2 = (const float*)d_in[9];
    const float* b2 = (const float*)d_in[10];
    const float* W3 = (const float*)d_in[11];
    const float* b3 = (const float*)d_in[12];
    const float* We = (const float*)d_in[13];
    const float* be = (const float*)d_in[14];
    const float* Wp = (const float*)d_in[15];
    const float* bp = (const float*)d_in[16];
    float* out = (float*)d_out;

    char* ws = (char*)d_ws;
    size_t off = 0;
    auto alloc = [&](size_t bytes) -> char* {
        char* p = ws + off;
        off = (off + bytes + 255) & ~(size_t)255;
        return p;
    };
    bf16* kpad = (bf16*)alloc((size_t)B_ROWS * KPAD * sizeof(bf16));
    bf16* W1t  = (bf16*)alloc((size_t)HDIM * KPAD * sizeof(bf16));
    bf16* W2t  = (bf16*)alloc((size_t)HDIM * HDIM * sizeof(bf16));
    bf16* W3t  = (bf16*)alloc((size_t)HDIM * HDIM * sizeof(bf16));
    bf16* Wpt  = (bf16*)alloc((size_t)NPOL * HDIM * sizeof(bf16));
    bf16* h1   = (bf16*)alloc((size_t)B_ROWS * HDIM * sizeof(bf16));
    bf16* h2   = (bf16*)alloc((size_t)B_ROWS * HDIM * sizeof(bf16));
    bf16* h3   = (bf16*)alloc((size_t)B_ROWS * HDIM * sizeof(bf16));
    float* lse = (float*)alloc((size_t)B_ROWS * sizeof(float));

    prep_k <<<(B_ROWS * KPAD + 255) / 256, 256, 0, stream>>>(pf, mb, ksf, mo, ps, kpad);
    prep_wt<<<(HDIM * KPAD + 255) / 256, 256, 0, stream>>>(W1, W1t, 135, KPAD, HDIM);
    prep_wt<<<(HDIM * HDIM + 255) / 256, 256, 0, stream>>>(W2, W2t, HDIM, HDIM, HDIM);
    prep_wt<<<(HDIM * HDIM + 255) / 256, 256, 0, stream>>>(W3, W3t, HDIM, HDIM, HDIM);
    prep_wt<<<(NPOL * HDIM + 255) / 256, 256, 0, stream>>>(Wp, Wpt, HDIM, HDIM, NPOL);

    gemm_bias_act<KPAD, true ><<<B_ROWS / 32, 256, 0, stream>>>(kpad, W1t, b1, h1);
    gemm_bias_act<HDIM, true ><<<B_ROWS / 32, 256, 0, stream>>>(h1,   W2t, b2, h2);
    gemm_bias_act<HDIM, false><<<B_ROWS / 32, 256, 0, stream>>>(h2,   W3t, b3, h3);

    logits_stats<<<B_ROWS / LSTAT_ROWS, 512, 0, stream>>>(h3, Wpt, bp, lse);
    eval_head   <<<B_ROWS / 8,  256, 0, stream>>>(h3, We, be, from_sq, to_sq,
                                                  out + (size_t)B_ROWS * 64);
    gather_probs<<<B_ROWS, 64, 0, stream>>>(h3, Wpt, bp, lse, from_sq, to_sq, out);
}
